// SeparableSpectralConv_21869973471499
// MI455X (gfx1250) — matrix operand order repaired and checked by reading, not compiled, not run on hardware
//
#include <hip/hip_runtime.h>
#include <hip/hip_bf16.h>

typedef __bf16 bf16;
typedef __attribute__((ext_vector_type(16))) __bf16 v16bf;
typedef __attribute__((ext_vector_type(8)))  __bf16 v8bf;
typedef __attribute__((ext_vector_type(16))) unsigned short v16us;
typedef __attribute__((ext_vector_type(8)))  float v8f;

// ---------- WMMA fragment helpers (CDNA5 wave32 layouts, CDNA5 ISA) ----------

// A fragment, 16x32 bf16, row-major source with leading dim lda.
// lane<16: M=lane, K = {0..7, 16..23}; lane>=16: M=lane-16, K = {8..15, 24..31}.
__device__ __forceinline__ v16bf load_a_frag(const bf16* A, int lda) {
  const int lane = threadIdx.x & 31;
  const bf16* p = A + (lane & 15) * lda + ((lane & 16) ? 8 : 0);
  v8bf lo = *(const v8bf*)p;
  v8bf hi = *(const v8bf*)(p + 16);
  return __builtin_shufflevector(lo, hi, 0,1,2,3,4,5,6,7,8,9,10,11,12,13,14,15);
}

// B fragment, 32x16 bf16, row-major source: N = lane&15, K = {0..7,16..23} (lane<16) / {8..15,24..31}.
__device__ __forceinline__ v16bf load_b_frag(const bf16* B, int ldb) {
  const int lane = threadIdx.x & 31;
  const bf16* p = B + (lane & 15) + (size_t)((lane & 16) ? 8 : 0) * ldb;
  v16bf b;
#pragma unroll
  for (int i = 0; i < 16; ++i) b[i] = p[(size_t)((i & 7) + ((i & 8) ? 16 : 0)) * ldb];
  return b;
}

// B fragment from f32 source (stage A streams X directly, converting to bf16).
__device__ __forceinline__ v16bf load_b_frag_f32(const float* B, int ldb) {
  const int lane = threadIdx.x & 31;
  const float* p = B + (lane & 15) + (size_t)((lane & 16) ? 8 : 0) * ldb;
  v16bf b;
#pragma unroll
  for (int i = 0; i < 16; ++i) b[i] = (bf16)p[(size_t)((i & 7) + ((i & 8) ? 16 : 0)) * ldb];
  return b;
}

__device__ __forceinline__ v16bf neg_frag(v16bf a) {
  v16us u = __builtin_bit_cast(v16us, a);
  u ^= (unsigned short)0x8000;
  return __builtin_bit_cast(v16bf, u);
}

__device__ __forceinline__ v8f wmma_bf16(v16bf a, v16bf b, v8f c) {
  return __builtin_amdgcn_wmma_f32_16x16x32_bf16(false, a, false, b, (short)0, c, false, false);
}

// ---------- problem constants ----------
// X: (16,128,128,128) f32 channels-last; kept modes: kx in {0..15}u{112..127} (32), ky 0..15.

// workspace offsets (bytes, all 256-aligned)
#define OFF_W1  0u          // [64 x 128]  bf16: fwd DFT dim1, rows 0..31 Re, 32..63 Im (x 1/128)
#define OFF_A2  16384u      // [32 x 256]  bf16: fwd DFT dim2 block [[Er,-Ei],[Ei,Er]] (x 1/128)
#define OFF_E1  32768u      // [256 x 64]  bf16: inv DFT dim1 block [[Er,-Ei],[Ei,Er]]
#define OFF_A3  65536u      // [128 x 32]  bf16: inv rfft dim2 [c*cos | -c*sin], c0=1 else 2
#define OFF_KR  73728u      // [32,16,128,128] bf16 kernel real
#define OFF_KI  (OFF_KR + 16777216u)          // kernel imag
#define OFF_T1  (OFF_KI + 16777216u)          // [16,2,32,128,128] bf16 (also aliased as T2)
#define OFF_XH  (OFF_T1 + 33554432u)          // [32,16,2,128,16]  bf16 (kx,ky,p,c,b)
#define OFF_YH  (OFF_XH + 4194304u)           // [16,2,32,16,128]  bf16 (b,p,kx,ky,o)
#define OFF_T2  OFF_T1                        // [16,128,2,16,128] bf16 (b,s1,p,ky,o)

// ---------- twiddle init ----------
__global__ void k_init_twiddles(bf16* W1, bf16* A2, bf16* E1, bf16* A3) {
  int idx = blockIdx.x * blockDim.x + threadIdx.x;
  const float w = 6.2831853071795864769f / 128.0f;
  if (idx < 8192) {                       // W1: 64 x 128
    int m = idx >> 7, s1 = idx & 127;
    int kx = m & 31;
    int f = (kx < 16) ? kx : kx + 96;
    float th = w * (float)((f * s1) & 127);
    float v = (m < 32) ? cosf(th) * (1.0f/128.0f) : -sinf(th) * (1.0f/128.0f);
    W1[idx] = (bf16)v;
  } else if (idx < 16384) {               // A2: 32 x 256
    int i2 = idx - 8192;
    int r = i2 >> 8, k = i2 & 255;
    int ky = r & 15, s2 = k & 127;
    float th = w * (float)((ky * s2) & 127);
    float c = cosf(th) * (1.0f/128.0f), s = sinf(th) * (1.0f/128.0f);
    float v = (r < 16) ? ((k < 128) ? c : s)      // [Er | -Ei], Ei = -s
                       : ((k < 128) ? -s : c);    // [Ei |  Er]
    A2[i2] = (bf16)v;
  } else if (idx < 32768) {               // E1: 256 x 64
    int i2 = idx - 16384;
    int m = i2 >> 6, r = i2 & 63;
    int kx = r & 31;
    int f = (kx < 16) ? kx : kx + 96;
    float th = w * (float)((f * (m & 127)) & 127);
    float c = cosf(th), s = sinf(th);
    float v = (m < 128) ? ((r < 32) ? c : -s)     // [E1r | -E1i]
                        : ((r < 32) ? s : c);     // [E1i |  E1r]
    E1[i2] = (bf16)v;
  } else if (idx < 36864) {               // A3: 128 x 32
    int i2 = idx - 32768;
    int s2 = i2 >> 5, k = i2 & 31;
    int ky = k & 15;
    float th = w * (float)((ky * s2) & 127);
    float cc = (ky == 0) ? 1.0f : 2.0f;
    float v = (k < 16) ? cc * cosf(th) : -cc * sinf(th);
    A3[i2] = (bf16)v;
  }
}

// ---------- kernel = sum_r k0*k1 (complex), kept modes only ----------
__global__ void k_combine(const float2* __restrict__ k0, const float2* __restrict__ k1,
                          bf16* __restrict__ Kr, bf16* __restrict__ Ki) {
  int idx = blockIdx.x * blockDim.x + threadIdx.x;   // ((kx*16+ky)*128+o)*128+i
  if (idx >= 512 * 16384) return;
  int i  = idx & 127;
  int o  = (idx >> 7) & 127;
  int ky = (idx >> 14) & 15;
  int kx = idx >> 18;
  size_t b0 = ((size_t)(kx * 128 + o) * 128 + i) * 4;   // k0: (32,1,128,128,4)
  size_t b1 = ((size_t)(ky * 128 + o) * 128 + i) * 4;   // k1: (1,17,128,128,4)
  float re = 0.f, im = 0.f;
#pragma unroll
  for (int r = 0; r < 4; ++r) {
    float2 a = k0[b0 + r], b = k1[b1 + r];
    re += a.x * b.x - a.y * b.y;
    im += a.x * b.y + a.y * b.x;
  }
  Kr[idx] = (bf16)re;
  Ki[idx] = (bf16)im;
}

// ---------- stage A: T1[b,p,kx,s2,c] = W1blk @ X[b]  (M=64,K=128,N=16384 per b) ----------
// One wave computes the entire M=64 strip for its 16-column slab: X fragment is
// loaded+converted once and feeds 4 WMMAs (X read exactly once from HBM).
__global__ void k_stageA(const float* __restrict__ X, const bf16* __restrict__ W1,
                         bf16* __restrict__ T1) {
  unsigned wid = blockIdx.x * 8u + (threadIdx.x >> 5);   // 16384 waves
  unsigned b  = wid >> 10;
  unsigned nt = wid & 1023;
  const float* Bb = X + (size_t)b * 2097152u + nt * 16;
  v8f acc[4] = {};
#pragma unroll
  for (int kk = 0; kk < 128; kk += 32) {
    __builtin_prefetch(Bb + (size_t)kk * 16384 + 8192, 0, 0);  // global_prefetch_b8
    v16bf bf = load_b_frag_f32(Bb + (size_t)kk * 16384, 16384);
#pragma unroll
    for (int mt = 0; mt < 4; ++mt)
      acc[mt] = wmma_bf16(load_a_frag(W1 + mt * 2048 + kk, 128), bf, acc[mt]);
  }
  const int lane = threadIdx.x & 31;
  const int n = nt * 16 + (lane & 15);
  const int mb = (lane & 16) ? 8 : 0;
#pragma unroll
  for (int mt = 0; mt < 4; ++mt)
#pragma unroll
    for (int r = 0; r < 8; ++r) {
      int m = mt * 16 + mb + r;                       // p = m>>5, kx = m&31
      unsigned row = (b * 2 + (m >> 5)) * 32 + (m & 31);
      T1[(size_t)row * 16384 + n] = (bf16)acc[mt][r];
    }
}

// ---------- stage B: Xh = A2blk @ [T1r;T1i]  (M=32,K=256,N=128 per (b,kx)) ----------
// One wave computes both M-tiles (Re+Im rows): T1 fragment read once, 2 WMMAs.
__global__ void k_stageB(const bf16* __restrict__ T1, const bf16* __restrict__ A2,
                         bf16* __restrict__ Xh) {
  unsigned wid = blockIdx.x * 8u + (threadIdx.x >> 5);   // 4096 waves
  unsigned batch = wid >> 3;                          // b*32+kx
  unsigned nt = wid & 7;
  unsigned b = batch >> 5, kx = batch & 31;
  v8f acc[2] = {};
#pragma unroll
  for (int kk = 0; kk < 256; kk += 32) {
    int p = kk >> 7, s2b = kk & 127;
    const bf16* Bb = T1 + (size_t)((b * 2 + p) * 32 + kx) * 16384 + s2b * 128 + nt * 16;
    v16bf bf = load_b_frag(Bb, 128);
#pragma unroll
    for (int mt = 0; mt < 2; ++mt)
      acc[mt] = wmma_bf16(load_a_frag(A2 + mt * 4096 + kk, 256), bf, acc[mt]);
  }
  const int lane = threadIdx.x & 31;
  const int c = nt * 16 + (lane & 15);
  const int mb = (lane & 16) ? 8 : 0;
#pragma unroll
  for (int mt = 0; mt < 2; ++mt)
#pragma unroll
    for (int r = 0; r < 8; ++r) {
      int m = mt * 16 + mb + r;                       // p = m>>4, ky = m&15
      unsigned idx = (((kx * 16 + (m & 15)) * 2 + (m >> 4)) * 128 + c) * 16 + b;
      Xh[idx] = (bf16)acc[mt][r];
    }
}

// ---------- stage C: Yh^T = Kc @ Xh^T (complex, M=128 o, K=128 i, N=16 b per mode) ----------
__global__ void k_stageC(const bf16* __restrict__ Xh, const bf16* __restrict__ Kr,
                         const bf16* __restrict__ Ki, bf16* __restrict__ Yh) {
  unsigned wid = blockIdx.x * 8u + (threadIdx.x >> 5);   // 4096 waves
  unsigned point = wid >> 3;                          // kx*16+ky
  unsigned mt = wid & 7;
  unsigned kx = point >> 4, ky = point & 15;
  const bf16* KrB = Kr + (size_t)point * 16384 + mt * 2048;
  const bf16* KiB = Ki + (size_t)point * 16384 + mt * 2048;
  const bf16* XrT = Xh + (size_t)point * 4096;        // (p=0,c,b), ldb=16
  const bf16* XiT = XrT + 2048;
  v8f accR = {}, accI = {};
#pragma unroll
  for (int kk = 0; kk < 128; kk += 32) {
    v16bf ar = load_a_frag(KrB + kk, 128);
    v16bf ai = load_a_frag(KiB + kk, 128);
    v16bf br = load_b_frag(XrT + kk * 16, 16);
    v16bf bi = load_b_frag(XiT + kk * 16, 16);
    accR = wmma_bf16(ar, br, accR);
    accR = wmma_bf16(neg_frag(ai), bi, accR);         // Re = Kr*Xr - Ki*Xi
    accI = wmma_bf16(ar, bi, accI);
    accI = wmma_bf16(ai, br, accI);                   // Im = Kr*Xi + Ki*Xr
  }
  const int lane = threadIdx.x & 31;
  const int bcol = lane & 15;
  const int mb = mt * 16 + ((lane & 16) ? 8 : 0);
#pragma unroll
  for (int r = 0; r < 8; ++r) {
    int o = mb + r;
    size_t base = ((size_t)(bcol * 2) * 32 + kx) * 2048 + ky * 128 + o;  // (b,p,kx,ky,o)
    Yh[base]             = (bf16)accR[r];
    Yh[base + 32 * 2048] = (bf16)accI[r];
  }
}

// ---------- stage D1: [T2r;T2i] = E1blk @ [Yhr;Yhi]  (M=256,K=64,N=2048 per b) ----------
// One wave computes a 4-tile M-strip: Yh fragment read once per K-step, 4 WMMAs.
__global__ void k_stageD1(const bf16* __restrict__ Yh, const bf16* __restrict__ E1,
                          bf16* __restrict__ T2) {
  unsigned wid = blockIdx.x * 8u + (threadIdx.x >> 5);   // 8192 waves
  unsigned b   = wid >> 9;
  unsigned mt4 = (wid >> 7) & 3;
  unsigned nt  = wid & 127;
  v8f acc[4] = {};
#pragma unroll
  for (int kk = 0; kk < 64; kk += 32) {
    const bf16* Bb = Yh + (size_t)((b * 2 + (kk >> 5)) * 32) * 2048 + nt * 16;
    v16bf bf = load_b_frag(Bb, 2048);
#pragma unroll
    for (int j = 0; j < 4; ++j)
      acc[j] = wmma_bf16(load_a_frag(E1 + (mt4 * 4 + j) * 1024 + kk, 64), bf, acc[j]);
  }
  const int lane = threadIdx.x & 31;
  const int n = nt * 16 + (lane & 15);
  const int mb = (lane & 16) ? 8 : 0;
#pragma unroll
  for (int j = 0; j < 4; ++j)
#pragma unroll
    for (int r = 0; r < 8; ++r) {
      int m = (mt4 * 4 + j) * 16 + mb + r;            // p = m>>7, s1 = m&127
      size_t idx = ((size_t)(b * 128 + (m & 127)) * 2 + (m >> 7)) * 2048 + n;
      T2[idx] = (bf16)acc[j][r];
    }
}

// ---------- stage D2: Y[b,s1] = A3 @ [T2r;T2i]  (M=128 s2, K=32, N=128 o per (b,s1)) ----------
// One wave computes the entire M=128 strip: T2 fragment read once, 8 WMMAs.
__global__ void k_stageD2(const bf16* __restrict__ T2, const bf16* __restrict__ A3,
                          float* __restrict__ Y) {
  unsigned wid = blockIdx.x * 8u + (threadIdx.x >> 5);   // 16384 waves
  unsigned bs = wid >> 3;                             // b*128+s1
  unsigned nt = wid & 7;
  v16bf bb = load_b_frag(T2 + (size_t)bs * 4096 + nt * 16, 128);
  v8f acc[8];
#pragma unroll
  for (int mt = 0; mt < 8; ++mt) {
    v8f z = {};
    acc[mt] = wmma_bf16(load_a_frag(A3 + mt * 512, 32), bb, z);
  }
  const int lane = threadIdx.x & 31;
  const int o = nt * 16 + (lane & 15);
  const int mb = (lane & 16) ? 8 : 0;
#pragma unroll
  for (int mt = 0; mt < 8; ++mt)
#pragma unroll
    for (int r = 0; r < 8; ++r)
      Y[((size_t)bs * 128 + (mt * 16 + mb + r)) * 128 + o] = acc[mt][r];
}

// ---------- launcher ----------
extern "C" void kernel_launch(void* const* d_in, const int* in_sizes, int n_in,
                              void* d_out, int out_size, void* d_ws, size_t ws_size,
                              hipStream_t stream) {
  (void)in_sizes; (void)n_in; (void)out_size; (void)ws_size;
  const float*  X  = (const float*)d_in[0];
  const float2* k0 = (const float2*)d_in[1];
  const float2* k1 = (const float2*)d_in[2];
  float* Y = (float*)d_out;
  char* ws = (char*)d_ws;

  bf16* W1 = (bf16*)(ws + OFF_W1);
  bf16* A2 = (bf16*)(ws + OFF_A2);
  bf16* E1 = (bf16*)(ws + OFF_E1);
  bf16* A3 = (bf16*)(ws + OFF_A3);
  bf16* Kr = (bf16*)(ws + OFF_KR);
  bf16* Ki = (bf16*)(ws + OFF_KI);
  bf16* T1 = (bf16*)(ws + OFF_T1);
  bf16* Xh = (bf16*)(ws + OFF_XH);
  bf16* Yh = (bf16*)(ws + OFF_YH);
  bf16* T2 = (bf16*)(ws + OFF_T2);   // aliases T1 (T1 dead after stage B)

  k_init_twiddles<<<144, 256, 0, stream>>>(W1, A2, E1, A3);
  k_combine<<<32768, 256, 0, stream>>>(k0, k1, Kr, Ki);
  k_stageA<<<2048, 256, 0, stream>>>(X, W1, T1);      // 16384 waves, M=64 strip each
  k_stageB<<<512, 256, 0, stream>>>(T1, A2, Xh);      // 4096 waves, M=32 strip each
  k_stageC<<<512, 256, 0, stream>>>(Xh, Kr, Ki, Yh);
  k_stageD1<<<1024, 256, 0, stream>>>(Yh, E1, T2);    // 8192 waves, 4-tile strip each
  k_stageD2<<<2048, 256, 0, stream>>>(T2, A3, Y);     // 16384 waves, M=128 strip each
}
